// RelativeGeometry_51238959841485
// MI455X (gfx1250) — matrix operand order repaired and checked by reading, not compiled, not run on hardware
//
#include <hip/hip_runtime.h>
#include <math.h>

typedef _Float16 v16h __attribute__((ext_vector_type(16)));
typedef float    v8f  __attribute__((ext_vector_type(8)));
typedef float    v2f  __attribute__((ext_vector_type(2)));

#define BB   4
#define NN   512
#define HID  128
#define OD   64
#define ROWS (BB*NN*NN)           // 1,048,576 pairs
#define TILES (ROWS/16)           // 65,536 16-row tiles
#define WPB  8                    // waves per block (256 threads, wave32)
#define NBLK 1024
#define ITERS (TILES/(NBLK*WPB))  // 8, exactly even

__global__ __launch_bounds__(256, 4)
void relgeom_mlp_kernel(const float* __restrict__ pos,
                        const float* __restrict__ color,
                        const float* __restrict__ psize,
                        const float* __restrict__ W1,
                        const float* __restrict__ b1,
                        const float* __restrict__ W2,
                        const float* __restrict__ b2,
                        float* __restrict__ out)
{
    __shared__ __align__(32) float    w1p[8*HID];   // W1 rows 0..6, row 7 = b1
    __shared__ __align__(32) _Float16 w2t[OD*HID];  // W2 transposed [n][k], f16
    __shared__ __align__(32) float    b2s[OD];

    const int tid = threadIdx.x;
    // ---- one-time weight staging (b1 folded into padded W1 row 7) ----
    for (int idx = tid; idx < 8*HID; idx += 256) {
        int k = idx >> 7, n = idx & 127;
        w1p[idx] = (k < 7) ? W1[k*HID + n] : b1[n];
    }
    for (int idx = tid; idx < HID*OD; idx += 256) {
        int k = idx >> 6, n = idx & 63;
        w2t[n*HID + k] = (_Float16)W2[idx];
    }
    if (tid < OD) b2s[tid] = b2[tid];
    __syncthreads();

    const int  lane   = tid & 31;
    const int  wave   = tid >> 5;
    const int  lrow   = lane & 15;       // pair index within tile / matrix row-col
    const bool hihalf = lane >= 16;
    const int  koff0  = hihalf ? 2 : 0;  // fp32 WMMA K base per lane-half
    const int  hoff   = hihalf ? 8 : 0;

    // ---- loop-invariant fragments in registers ----
    // Layer-1 A-fragments: W1^T tiles (rows = hidden units), 16x4 f32 layout
    v2f w1f[8][2];
    #pragma unroll
    for (int t8 = 0; t8 < 8; ++t8) {
        const int n = t8*16 + lrow;
        #pragma unroll
        for (int ks = 0; ks < 2; ++ks) {
            w1f[t8][ks][0] = w1p[(ks*4 + koff0    )*HID + n];
            w1f[t8][ks][1] = w1p[(ks*4 + koff0 + 1)*HID + n];
        }
    }
    // Layer-2 B-fragments: W2 32x16 f16 tiles (feed the back-to-back WMMA burst)
    v16h w2f[4][4];
    #pragma unroll
    for (int s = 0; s < 4; ++s)
        #pragma unroll
        for (int t = 0; t < 4; ++t) {
            const _Float16* wp = w2t + (t*16 + lrow)*HID + s*32 + (hihalf ? 8 : 0);
            #pragma unroll
            for (int e = 0; e < 16; ++e)
                w2f[s][t][e] = wp[e < 8 ? e : e + 8];
        }
    float b2v[4];
    #pragma unroll
    for (int t = 0; t < 4; ++t) b2v[t] = b2s[t*16 + lrow];

    for (int it = 0; it < ITERS; ++it) {
        const int tile = blockIdx.x*WPB + wave + it*(NBLK*WPB);
        const int m0   = tile << 4;
        const int m    = m0 + lrow;
        const int b    = m >> 18;                  // N*N = 2^18
        const int ij   = m & (NN*NN - 1);
        const int i    = ij >> 9;
        const int j    = ij & (NN - 1);

        // ---- geometry features (8th feature = 1.0 -> adds b1 via W1 row 7) ----
        const float* pb = pos   + (size_t)b*NN*2;
        const float* cb = color + (size_t)b*NN*3;
        const float* sb = psize + (size_t)b*NN;
        const float dxv = pb[2*i]   - pb[2*j];
        const float dyv = pb[2*i+1] - pb[2*j+1];
        const float g0 = sqrtf(dxv*dxv + dyv*dyv + 1e-6f);
        const float g1 = dxv;
        const float g2 = dyv;
        const float g3 = cb[3*i]   - cb[3*j];
        const float g4 = cb[3*i+1] - cb[3*j+1];
        const float g5 = cb[3*i+2] - cb[3*j+2];
        const float g6 = sb[i] - sb[j];

        // Layer-1 B-fragments: geom^T (4 x 16 pairs), per-lane-half K selection
        v2f gf0, gf1;
        gf0[0] = hihalf ? g2 : g0;   gf0[1] = hihalf ? g3   : g1;
        gf1[0] = hihalf ? g6 : g4;   gf1[1] = hihalf ? 1.0f : g5;  // K7 = 1 (bias)

        // ---- output accumulators, init with b2 (N const per lane) ----
        v8f acc2[4];
        #pragma unroll
        for (int t = 0; t < 4; ++t)
            #pragma unroll
            for (int r = 0; r < 8; ++r) acc2[t][r] = b2v[t];

        // ---- fused layer1(transposed) -> GELU -> pack -> layer2, per 32-K step
        #pragma unroll
        for (int s = 0; s < 4; ++s) {
            v16h afrag;
            #pragma unroll
            for (int sub = 0; sub < 2; ++sub) {
                const int t8 = 2*s + sub;
                // h^T tile: rows = hidden units 16*t8..+15, cols = 16 pairs
                v8f acc = (v8f)(0.0f);
                acc = __builtin_amdgcn_wmma_f32_16x16x4_f32(
                          false, w1f[t8][0], false, gf0, (short)0, acc, false, false);
                acc = __builtin_amdgcn_wmma_f32_16x16x4_f32(
                          false, w1f[t8][1], false, gf1, (short)0, acc, false, false);
                // exact GELU; lane now holds h[K][own pair row] in f16 A-frag order
                #pragma unroll
                for (int r = 0; r < 8; ++r) {
                    const float x   = acc[r];
                    const float gel = 0.5f * x * (1.0f + erff(x * 0.70710678118654752f));
                    afrag[sub*8 + r] = (_Float16)gel;
                }
            }
            #pragma unroll
            for (int t = 0; t < 4; ++t)
                acc2[t] = __builtin_amdgcn_wmma_f32_16x16x32_f16(
                              false, afrag, false, w2f[s][t], (short)0, acc2[t],
                              false, false);
        }

        // ---- store 16x64 f32 output tile ----
        float* orow = out + (size_t)m0 * OD;
        #pragma unroll
        for (int t = 0; t < 4; ++t) {
            const int col = t*16 + lrow;
            #pragma unroll
            for (int r = 0; r < 8; ++r)
                orow[(size_t)(r + hoff)*OD + col] = acc2[t][r];
        }
    }
}

extern "C" void kernel_launch(void* const* d_in, const int* in_sizes, int n_in,
                              void* d_out, int out_size, void* d_ws, size_t ws_size,
                              hipStream_t stream) {
    (void)in_sizes; (void)n_in; (void)d_ws; (void)ws_size; (void)out_size;
    const float* pos   = (const float*)d_in[0];
    const float* color = (const float*)d_in[1];
    const float* psize = (const float*)d_in[2];
    const float* W1    = (const float*)d_in[3];
    const float* b1    = (const float*)d_in[4];
    const float* W2    = (const float*)d_in[5];
    const float* b2    = (const float*)d_in[6];
    float* out = (float*)d_out;
    relgeom_mlp_kernel<<<NBLK, 256, 0, stream>>>(pos, color, psize, W1, b1, W2, b2, out);
}
